// _ParallelBlock_7919919694497
// MI455X (gfx1250) — matrix operand order repaired and checked by reading, not compiled, not run on hardware
//
#include <hip/hip_runtime.h>
#include <hip/hip_bf16.h>
#include <math.h>
#include <stdint.h>

typedef __attribute__((ext_vector_type(16))) _Float16 v16h;
typedef __attribute__((ext_vector_type(8)))  float    v8f;
typedef __attribute__((address_space(3))) const _Float16 lds_cf16;

#define D_MODEL  512
#define N_HEADS  32
#define HEAD_DIM 16
#define D_INNER  1024
#define D_STATE  64
#define HDIM     64
#define NH       16
#define D_CONV   4
#define CONV_CH  (D_INNER + 2*D_STATE)          /* 1152 */
#define D_INPROJ (2*D_INNER + 2*D_STATE + NH)   /* 2192 */
#define BB 4
#define LL 1024
#define NT (BB*LL)

// ---------------------------------------------------- CDNA5 async helpers
__device__ __forceinline__ void async_copy_b128(uint32_t lds_off, const void* g) {
  // each enabled lane copies 16B global -> LDS[lds_off] (ASYNCcnt-tracked)
  asm volatile("global_load_async_to_lds_b128 %0, %1, off"
               :: "v"(lds_off), "v"((uint64_t)(uintptr_t)g) : "memory");
}
template <int N>
__device__ __forceinline__ void wait_async() {
  asm volatile("s_wait_asynccnt %0" :: "i"(N) : "memory");
}

// ---------------------------------------------------------------- converter
__global__ void k_f32_to_f16(const float* __restrict__ src,
                             _Float16* __restrict__ dst, int n) {
  int i = blockIdx.x * 256 + threadIdx.x;
  if (i < n) dst[i] = (_Float16)src[i];
}

// ---------------------------------------------------------------- WMMA GEMM
// C[M,N] = A[M,K](f16 rm) x W[N,K]^T(f16 rm) + bias.
// 256 threads = 8 waves; macro-tile 128(M) x 64(N); BK=32 double-buffered in
// LDS via async global->LDS copies. Wave w owns rows [16w,16w+16) x all 64 cols
// (1 A-frag reused across 4 B-frags -> 4 WMMAs per K-step).
#define BM 128
#define BN 64
#define BK 32
#define A_BYTES (BM*BK*2)                 /* 8192 */
#define B_BYTES (BN*BK*2)                 /* 4096 */
#define STAGE_BYTES (A_BYTES + B_BYTES)   /* 12288 */

__global__ void k_gemm_f16(const _Float16* __restrict__ A,
                           const _Float16* __restrict__ W,
                           const float* __restrict__ bias,
                           float* __restrict__ C, int N, int K) {
  __shared__ __align__(16) char lds[2 * STAGE_BYTES];
  // Launder the LDS base through asm: (1) escapes/captures the buffer so the
  // async-to-LDS writes (asm "memory") are assumed to alias it and the ds
  // reads are NOT folded to undef; (2) low 32 bits of the flat address are
  // the LDS byte offset,
  // exactly what global_load_async_to_lds wants in VDST.
  uint64_t flat = (uint64_t)(uintptr_t)&lds[0];
  asm volatile("" : "+v"(flat));
  const uint32_t lbase = (uint32_t)flat;

  const int tid  = threadIdx.x;
  const int lane = tid & 31, wv = tid >> 5;
  const int lo = lane & 15, hi = lane >> 4;
  const int m0 = blockIdx.x * BM;
  const int n0 = blockIdx.y * BN;

  // stage filler: 16B chunks; A = 512 chunks (2/thread), B = 256 chunks (1/thread)
  auto issue = [&](int k0, int st) {
    const uint32_t base = lbase + (uint32_t)(st * STAGE_BYTES);
    int c = tid;                                   // A chunk 0
    async_copy_b128(base + c * 16,
                    A + (size_t)(m0 + (c >> 2)) * K + k0 + (c & 3) * 8);
    c = tid + 256;                                 // A chunk 1
    async_copy_b128(base + c * 16,
                    A + (size_t)(m0 + (c >> 2)) * K + k0 + (c & 3) * 8);
    int wr = n0 + (tid >> 2);                      // B chunk (clamp ragged N)
    if (wr >= N) wr = N - 1;
    async_copy_b128(base + A_BYTES + tid * 16,
                    W + (size_t)wr * K + k0 + (tid & 3) * 8);
  };

  v8f acc[4];
  #pragma unroll
  for (int ns = 0; ns < 4; ++ns) acc[ns] = (v8f){};

  const int nk = K / BK;
  issue(0, 0);
  for (int i = 0; i < nk; ++i) {
    const int st = i & 1;
    if (i + 1 < nk) { issue((i + 1) * BK, st ^ 1); wait_async<3>(); }
    else            { wait_async<0>(); }
    __syncthreads();
    const lds_cf16* smA = (const lds_cf16*)(lbase + (uint32_t)(st * STAGE_BYTES));
    const lds_cf16* smB = (const lds_cf16*)(lbase + (uint32_t)(st * STAGE_BYTES + A_BYTES));
    // A fragment (16x32): lanes 0-15 M=lo; elem e -> K=hi*16+e
    v16h a;
    const lds_cf16* ar = smA + (wv * 16 + lo) * BK;
    #pragma unroll
    for (int j = 0; j < 8; ++j) {
      a[j]     = ar[hi * 16 + j];
      a[8 + j] = ar[hi * 16 + 8 + j];
    }
    #pragma unroll
    for (int ns = 0; ns < 4; ++ns) {
      // B fragment (32x16): lane n=lo, elem e -> K = hi*16+e
      v16h b;
      const lds_cf16* br = smB + (ns * 16 + lo) * BK + hi * 16;
      #pragma unroll
      for (int e = 0; e < 16; ++e) b[e] = br[e];
      acc[ns] = __builtin_amdgcn_wmma_f32_16x16x32_f16(false, a, false, b,
                                                       (short)0, acc[ns],
                                                       false, false);
    }
    __syncthreads();
  }

  #pragma unroll
  for (int ns = 0; ns < 4; ++ns) {
    int col = n0 + ns * 16 + lo;
    if (col < N) {
      float bv = bias ? bias[col] : 0.f;
      #pragma unroll
      for (int r = 0; r < 8; ++r) {
        int row = m0 + wv * 16 + hi * 8 + r;
        C[(size_t)row * N + col] = acc[ns][r] + bv;
      }
    }
  }
}

// ---------------------------------------------------------------- attention
// one block per (q-tile of 16, head, batch); 128 threads (4 waves)
// dyn LDS: 16*1024 f32 scores + 4*256 f32 reduction
__global__ void k_attn(const _Float16* __restrict__ qkv,
                       _Float16* __restrict__ o16) {
  extern __shared__ float smem[];
  float* sc  = smem;             // [16][1024]
  float* red = smem + 16 * LL;   // [4][256]
  const int lane = threadIdx.x & 31, wv = threadIdx.x >> 5;
  const int lo = lane & 15, hi = lane >> 4;
  const int q0 = blockIdx.x * 16, hh = blockIdx.y, b = blockIdx.z;

  // Q fragment: head-dim 16 zero-padded to K=32
  v16h aq = {};
  const _Float16* Qb = qkv + (size_t)(b * LL + q0 + lo) * (3 * D_MODEL)
                           + hh * HEAD_DIM;
  #pragma unroll
  if (hi == 0) for (int j = 0; j < 16; ++j) aq[j] = Qb[j];   // K=j < 16 (hi==0)

  // ---- scores ----
  for (int i = 0; i < 16; ++i) {
    int k0 = (wv * 16 + i) * 16;                         // key-tile base
    v16h bk = {};
    if (hi == 0) {                                       // K(dim) = e < 16
      const _Float16* Kb = qkv + (size_t)(b * LL + k0 + lo) * (3 * D_MODEL)
                               + D_MODEL + hh * HEAD_DIM;
      #pragma unroll
      for (int e = 0; e < 16; ++e) bk[e] = Kb[e];
    }
    v8f s = {};
    s = __builtin_amdgcn_wmma_f32_16x16x32_f16(false, aq, false, bk,
                                               (short)0, s, false, false);
    #pragma unroll
    for (int r = 0; r < 8; ++r)
      sc[(hi * 8 + r) * LL + k0 + lo] = s[r] * 0.25f;    // 1/sqrt(16)
  }
  __syncthreads();

  // ---- softmax: 8 threads per row ----
  {
    const int row = threadIdx.x >> 3, j8 = threadIdx.x & 7;
    float* srow = sc + row * LL;
    float mx = -1e30f;
    for (int c = j8; c < LL; c += 8) mx = fmaxf(mx, srow[c]);
    #pragma unroll
    for (int d = 1; d < 8; d <<= 1) mx = fmaxf(mx, __shfl_xor(mx, d, 8));
    float sum = 0.f;
    for (int c = j8; c < LL; c += 8) {
      float e = __expf(srow[c] - mx); srow[c] = e; sum += e;
    }
    #pragma unroll
    for (int d = 1; d < 8; d <<= 1) sum += __shfl_xor(sum, d, 8);
    float inv = 1.f / sum;
    for (int c = j8; c < LL; c += 8) srow[c] *= inv;
  }
  __syncthreads();

  // ---- probs x V : each wave handles 256 keys ----
  v8f acc = {};
  for (int c0 = wv * 256; c0 < (wv + 1) * 256; c0 += 32) {
    v16h ap, bv;
    const float* pr = sc + lo * LL + c0;                 // A row m = q = lo
    #pragma unroll
    for (int e = 0; e < 8; ++e) {
      ap[e]     = (_Float16)pr[hi * 16 + e];
      ap[8 + e] = (_Float16)pr[hi * 16 + 8 + e];
    }
    const _Float16* Vb = qkv + (size_t)(b * LL + c0 + hi * 16) * (3 * D_MODEL)
                             + 2 * D_MODEL + hh * HEAD_DIM + lo;
    #pragma unroll
    for (int e = 0; e < 16; ++e) bv[e] = Vb[(size_t)e * (3 * D_MODEL)];
    acc = __builtin_amdgcn_wmma_f32_16x16x32_f16(false, ap, false, bv,
                                                 (short)0, acc, false, false);
  }
  #pragma unroll
  for (int r = 0; r < 8; ++r) red[wv * 256 + (hi * 8 + r) * 16 + lo] = acc[r];
  __syncthreads();
  for (int idx = threadIdx.x; idx < 256; idx += 128) {
    float s = red[idx] + red[256 + idx] + red[512 + idx] + red[768 + idx];
    int r = idx >> 4, c = idx & 15;
    o16[(size_t)(b * LL + q0 + r) * D_MODEL + hh * HEAD_DIM + c] = (_Float16)s;
  }
}

// ----------------------------------------------------- causal dwconv + silu
__global__ void k_conv_silu(const float* __restrict__ zx,
                            const float* __restrict__ cw,
                            const float* __restrict__ cb,
                            float* __restrict__ xbc, int rev) {
  int i = blockIdx.x * 256 + threadIdx.x;
  if (i >= NT * CONV_CH) return;
  int c = i % CONV_CH, bt = i / CONV_CH;
  int t = bt % LL, b = bt / LL;
  float acc = cb[c];
  #pragma unroll
  for (int j = 0; j < D_CONV; ++j) {
    int tau = t - (D_CONV - 1) + j;
    if (tau >= 0) {
      int src = rev ? (LL - 1 - tau) : tau;
      acc += cw[c * D_CONV + j] *
             zx[(size_t)(b * LL + src) * D_INPROJ + D_INNER + c];
    }
  }
  xbc[i] = acc / (1.f + __expf(-acc));   // silu
}

// ---------------------------------------------------------------- SSD scan
__global__ void k_scan(const float* __restrict__ zx,
                       const float* __restrict__ xbc,
                       const float* __restrict__ dtb,
                       const float* __restrict__ alog,
                       const float* __restrict__ Dpar,
                       float* __restrict__ y, int rev) {
  __shared__ float xs[HDIM], Bs[D_STATE], Cs[D_STATE];
  const int b = blockIdx.x >> 4, n = blockIdx.x & 15;
  const int tid = threadIdx.x;
  const float Ac = -__expf(alog[n]);
  const float biasdt = dtb[n], Dn = Dpar[n];
  const int p = tid >> 2, so = (tid & 3) * 16;
  float st[16];
  #pragma unroll
  for (int j = 0; j < 16; ++j) st[j] = 0.f;
  for (int t = 0; t < LL; ++t) {
    const size_t base = (size_t)(b * LL + t) * CONV_CH;
    if (tid < 64)       xs[tid]       = xbc[base + n * HDIM + tid];
    else if (tid < 128) Bs[tid - 64]  = xbc[base + D_INNER + (tid - 64)];
    else if (tid < 192) Cs[tid - 128] = xbc[base + D_INNER + D_STATE + (tid - 128)];
    __syncthreads();
    const int src = rev ? (LL - 1 - t) : t;
    float draw = zx[(size_t)(b * LL + src) * D_INPROJ + (D_INNER + CONV_CH) + n]
                 + biasdt;
    float dt = (draw > 20.f) ? draw : log1pf(__expf(draw));   // softplus
    float dA = __expf(dt * Ac);
    float dtx = dt * xs[p];
    float yp = 0.f;
    #pragma unroll
    for (int j = 0; j < 16; ++j) {
      st[j] = dA * st[j] + dtx * Bs[so + j];
      yp += st[j] * Cs[so + j];
    }
    yp += __shfl_xor(yp, 1);
    yp += __shfl_xor(yp, 2);
    if ((tid & 3) == 0)
      y[(size_t)(b * LL + src) * D_INNER + n * HDIM + p] = yp + Dn * xs[p];
    __syncthreads();
  }
}

// ------------------------------------------------- silu-gated RMSNorm -> f16
__global__ void k_gated_rmsnorm(const float* __restrict__ y,
                                const float* __restrict__ zx,
                                const float* __restrict__ nw,
                                _Float16* __restrict__ y16) {
  __shared__ float rbuf[256];
  const int tok = blockIdx.x, tid = threadIdx.x;
  float v[4]; float ss = 0.f;
  #pragma unroll
  for (int i = 0; i < 4; ++i) {
    int d = tid + i * 256;
    float z = zx[(size_t)tok * D_INPROJ + d];
    float val = y[(size_t)tok * D_INNER + d] * (z / (1.f + __expf(-z)));
    v[i] = val; ss += val * val;
  }
  rbuf[tid] = ss; __syncthreads();
  for (int s = 128; s > 0; s >>= 1) {
    if (tid < s) rbuf[tid] += rbuf[tid + s];
    __syncthreads();
  }
  float r = rsqrtf(rbuf[0] / (float)D_INNER + 1e-5f);
  #pragma unroll
  for (int i = 0; i < 4; ++i) {
    int d = tid + i * 256;
    y16[(size_t)tok * D_INNER + d] = (_Float16)(v[i] * r * nw[d]);
  }
}

// ------------------------------------------------- residual sum + LayerNorm
__global__ void k_final_ln(const float* __restrict__ h,
                           const float* __restrict__ at,
                           const float* __restrict__ fw,
                           const float* __restrict__ bw,
                           const float* __restrict__ lnw,
                           const float* __restrict__ lnb,
                           float* __restrict__ out) {
  __shared__ float r1[256], r2[256];
  const int tok = blockIdx.x, tid = threadIdx.x;
  const size_t base = (size_t)tok * D_MODEL;
  float s0 = h[base + tid] + at[base + tid] + fw[base + tid] + bw[base + tid];
  float s1 = h[base + tid + 256] + at[base + tid + 256] +
             fw[base + tid + 256] + bw[base + tid + 256];
  r1[tid] = s0 + s1; r2[tid] = s0 * s0 + s1 * s1;
  __syncthreads();
  for (int s = 128; s > 0; s >>= 1) {
    if (tid < s) { r1[tid] += r1[tid + s]; r2[tid] += r2[tid + s]; }
    __syncthreads();
  }
  float mu  = r1[0] / (float)D_MODEL;
  float var = r2[0] / (float)D_MODEL - mu * mu;
  float inv = rsqrtf(var + 1e-5f);
  out[base + tid]       = (s0 - mu) * inv * lnw[tid]       + lnb[tid];
  out[base + tid + 256] = (s1 - mu) * inv * lnw[tid + 256] + lnb[tid + 256];
}

// ------------------------------------------------------------------- launch
extern "C" void kernel_launch(void* const* d_in, const int* in_sizes, int n_in,
                              void* d_out, int out_size, void* d_ws, size_t ws_size,
                              hipStream_t stream) {
  (void)in_sizes; (void)n_in; (void)out_size;
  const float* h          = (const float*)d_in[0];
  const float* attn_in_w  = (const float*)d_in[1];
  const float* attn_in_b  = (const float*)d_in[2];
  const float* attn_out_w = (const float*)d_in[3];
  const float* attn_out_b = (const float*)d_in[4];
  const float* inproj_w[2]  = {(const float*)d_in[5],  (const float*)d_in[13]};
  const float* conv_w[2]    = {(const float*)d_in[6],  (const float*)d_in[14]};
  const float* conv_b[2]    = {(const float*)d_in[7],  (const float*)d_in[15]};
  const float* dt_bias[2]   = {(const float*)d_in[8],  (const float*)d_in[16]};
  const float* A_log[2]     = {(const float*)d_in[9],  (const float*)d_in[17]};
  const float* Dpar[2]      = {(const float*)d_in[10], (const float*)d_in[18]};
  const float* norm_w[2]    = {(const float*)d_in[11], (const float*)d_in[19]};
  const float* outproj_w[2] = {(const float*)d_in[12], (const float*)d_in[20]};
  const float* ln_w = (const float*)d_in[21];
  const float* ln_b = (const float*)d_in[22];
  float* out = (float*)d_out;

  // ---- workspace layout (bytes) ----
  constexpr size_t SZ_H16    = (size_t)NT * D_MODEL  * 2;
  constexpr size_t SZ_W16    = (size_t)D_INPROJ * D_MODEL * 2;
  constexpr size_t SZ_TOK512 = (size_t)NT * D_MODEL  * 4;
  constexpr size_t OFF_S     = SZ_H16 + SZ_W16 + 3 * SZ_TOK512;
  constexpr size_t SZ_QKV32  = (size_t)NT * 3 * D_MODEL * 4;
  constexpr size_t SZ_QKV16  = (size_t)NT * 3 * D_MODEL * 2;
  constexpr size_t SZ_O16    = (size_t)NT * D_MODEL * 2;
  constexpr size_t SZ_ZX     = (size_t)NT * D_INPROJ * 4;
  constexpr size_t SZ_XBC    = (size_t)NT * CONV_CH * 4;
  constexpr size_t SZ_Y32    = (size_t)NT * D_INNER * 4;
  constexpr size_t SZ_Y16    = (size_t)NT * D_INNER * 2;
  constexpr size_t SZ_ATTN_S = SZ_QKV32 + SZ_QKV16 + SZ_O16;
  constexpr size_t SZ_MAMBA_S= SZ_ZX + SZ_XBC + SZ_Y32 + SZ_Y16;
  constexpr size_t NEEDED = OFF_S + (SZ_ATTN_S > SZ_MAMBA_S ? SZ_ATTN_S : SZ_MAMBA_S);
  if (ws_size < NEEDED) return;   // cannot run without scratch

  char* ws = (char*)d_ws;
  _Float16* h16    = (_Float16*)(ws);
  _Float16* w16    = (_Float16*)(ws + SZ_H16);
  float*    attn32 = (float*)(ws + SZ_H16 + SZ_W16);
  float*    dir32[2] = {(float*)(ws + SZ_H16 + SZ_W16 + SZ_TOK512),
                        (float*)(ws + SZ_H16 + SZ_W16 + 2 * SZ_TOK512)};
  char* S = ws + OFF_S;
  // attention-phase aliases
  float*    qkv32 = (float*)S;
  _Float16* qkv16 = (_Float16*)(S + SZ_QKV32);
  _Float16* o16   = (_Float16*)(S + SZ_QKV32 + SZ_QKV16);
  // mamba-phase aliases (same region; attention finished by then)
  float*    zx32  = (float*)S;
  float*    xbc32 = (float*)(S + SZ_ZX);
  float*    y32   = (float*)(S + SZ_ZX + SZ_XBC);
  _Float16* y16   = (_Float16*)(S + SZ_ZX + SZ_XBC + SZ_Y32);

  auto cvt = [&](const float* src, _Float16* dst, int n) {
    k_f32_to_f16<<<(n + 255) / 256, 256, 0, stream>>>(src, dst, n);
  };

  cvt(h, h16, NT * D_MODEL);

  // ---------------- attention ----------------
  cvt(attn_in_w, w16, 3 * D_MODEL * D_MODEL);
  k_gemm_f16<<<dim3(NT / BM, (3 * D_MODEL) / BN), 256, 0, stream>>>(
      h16, w16, attn_in_b, qkv32, 3 * D_MODEL, D_MODEL);
  cvt(qkv32, qkv16, NT * 3 * D_MODEL);
  size_t smem = (size_t)(16 * LL + 4 * 256) * sizeof(float);   // 69.6 KB
  k_attn<<<dim3(LL / 16, N_HEADS, BB), 128, smem, stream>>>(qkv16, o16);
  cvt(attn_out_w, w16, D_MODEL * D_MODEL);
  k_gemm_f16<<<dim3(NT / BM, D_MODEL / BN), 256, 0, stream>>>(
      o16, w16, attn_out_b, attn32, D_MODEL, D_MODEL);

  // ---------------- mamba fwd (rev=0) / bwd (rev=1) ----------------
  for (int d = 0; d < 2; ++d) {
    cvt(inproj_w[d], w16, D_INPROJ * D_MODEL);
    k_gemm_f16<<<dim3(NT / BM, (D_INPROJ + BN - 1) / BN), 256, 0, stream>>>(
        h16, w16, nullptr, zx32, D_INPROJ, D_MODEL);
    k_conv_silu<<<(NT * CONV_CH + 255) / 256, 256, 0, stream>>>(
        zx32, conv_w[d], conv_b[d], xbc32, d);
    k_scan<<<BB * NH, 256, 0, stream>>>(
        zx32, xbc32, dt_bias[d], A_log[d], Dpar[d], y32, d);
    k_gated_rmsnorm<<<NT, 256, 0, stream>>>(y32, zx32, norm_w[d], y16);
    cvt(outproj_w[d], w16, D_MODEL * D_INNER);
    k_gemm_f16<<<dim3(NT / BM, D_MODEL / BN), 256, 0, stream>>>(
        y16, w16, nullptr, dir32[d], D_MODEL, D_INNER);
  }

  // ---------------- residual + LayerNorm ----------------
  k_final_ln<<<NT, 256, 0, stream>>>(h, attn32, dir32[0], dir32[1],
                                     ln_w, ln_b, out);
}
